// DiscoBertModel_58394375356713
// MI455X (gfx1250) — matrix operand order repaired and checked by reading, not compiled, not run on hardware
//
#include <hip/hip_runtime.h>
#include <hip/hip_bf16.h>
#include <math.h>

typedef __bf16 v16bf __attribute__((ext_vector_type(16)));
typedef __bf16 v4bf  __attribute__((ext_vector_type(4)));
typedef float  v8f   __attribute__((ext_vector_type(8)));

#define N_EDUS 512
#define SEQ    128
#define SROWS  127   // tokens after dropping [CLS]
#define HB     768
#define A_DIM  100
#define A_PAD  112   // 7 * 16
#define H_DIM  256
#define HALF   128
#define KC     64
#define NKC    (HB / KC)   // 12

#if defined(__has_builtin)
#if __has_builtin(__builtin_amdgcn_sched_barrier)
#define SCHED_FENCE() __builtin_amdgcn_sched_barrier(0)
#endif
#endif
#ifndef SCHED_FENCE
#define SCHED_FENCE() ((void)0)
#endif

// ---- WMMA fragment swizzle (CDNA5 ISA §7.12.2, 16-bit layouts) ----
// A (16x32 bf16): lane<16 -> M=lane, V0..3: K=0..7, V4..7: K=16..23
//                 lane>=16 -> M=lane-16, V0..3: K=8..15, V4..7: K=24..31
__device__ __forceinline__ int aswz(int s, int k) {
  int mt = s >> 4, m = s & 15;
  int ks = k >> 5, kk = k & 31;
  int lane = m + (((kk >> 3) & 1) << 4);
  int v = ((kk >> 4) << 2) + ((kk & 7) >> 1);
  int e = (v << 1) + (kk & 1);
  return ((mt * 2 + ks) * 32 + lane) * 16 + e;
}
// B (32x16 bf16): lanes 0-15 hold K=0..7,16..23 (N=lane), lanes 16-31 hold K=8..15,24..31
__device__ __forceinline__ int bswz(int k, int a) {
  int nt = a >> 4, n = a & 15;
  int ks = k >> 5, kk = k & 31;
  int lane = n + (((kk >> 3) & 1) << 4);
  int v = ((kk >> 4) << 2) + ((kk & 7) >> 1);
  int e = (v << 1) + (kk & 1);
  return ((nt * 2 + ks) * 32 + lane) * 16 + e;
}

// ---- software-pipeline helpers -------------------------------------------
__device__ __forceinline__ void load_regs(const float* __restrict__ seqn,
                                          const float* __restrict__ w1,
                                          int kbase, int t,
                                          float4* ra, float4* rb) {
  #pragma unroll
  for (int i = 0; i < 8; ++i) {            // A: 128x64, float4 per thread x8
    int idx = i * 256 + t;
    int s = idx >> 4, k4 = (idx & 15) << 2;
    ra[i] = make_float4(0.f, 0.f, 0.f, 0.f);
    if (s < SROWS) ra[i] = *reinterpret_cast<const float4*>(&seqn[s * HB + kbase + k4]);
  }
  #pragma unroll
  for (int i = 0; i < 7; ++i) {            // B: 64x112 (pad from 100)
    int idx = i * 256 + t;
    int k = idx / 28, a4 = (idx - k * 28) << 2;
    rb[i] = make_float4(0.f, 0.f, 0.f, 0.f);
    if (a4 + 3 < A_DIM)                    // 400B rows stay 16B-aligned
      rb[i] = *reinterpret_cast<const float4*>(&w1[(kbase + k) * A_DIM + a4]);
  }
}

__device__ __forceinline__ void store_lds(__bf16* __restrict__ A,
                                          __bf16* __restrict__ B, int t,
                                          const float4* ra, const float4* rb) {
  #pragma unroll
  for (int i = 0; i < 8; ++i) {
    int idx = i * 256 + t;
    int s = idx >> 4, k4 = (idx & 15) << 2;
    v4bf p = { (__bf16)ra[i].x, (__bf16)ra[i].y, (__bf16)ra[i].z, (__bf16)ra[i].w };
    *reinterpret_cast<v4bf*>(&A[aswz(s, k4)]) = p;   // 4 consecutive slots -> b64
  }
  #pragma unroll
  for (int i = 0; i < 7; ++i) {
    int idx = i * 256 + t;
    int k = idx / 28, a4 = (idx - k * 28) << 2;
    B[bswz(k, a4 + 0)] = (__bf16)rb[i].x;
    B[bswz(k, a4 + 1)] = (__bf16)rb[i].y;
    B[bswz(k, a4 + 2)] = (__bf16)rb[i].z;
    B[bswz(k, a4 + 3)] = (__bf16)rb[i].w;
  }
}

__device__ __forceinline__ void compute_chunk(const __bf16* __restrict__ A,
                                              const __bf16* __restrict__ B,
                                              int lane, int wv, v8f* acc) {
  #pragma unroll
  for (int ks = 0; ks < 2; ++ks) {
    v16bf af = *reinterpret_cast<const v16bf*>(&A[((wv * 2 + ks) * 32 + lane) * 16]);
    v16bf bfr[7];
    #pragma unroll
    for (int nt = 0; nt < 7; ++nt)
      bfr[nt] = *reinterpret_cast<const v16bf*>(&B[((nt * 2 + ks) * 32 + lane) * 16]);
    SCHED_FENCE();                          // all DS loads before any WMMA
    #pragma unroll
    for (int nt = 0; nt < 7; ++nt)
      acc[nt] = __builtin_amdgcn_wmma_f32_16x16x32_bf16(
          false, af, false, bfr[nt], (short)0, acc[nt], false, false);
  }
}

// =========================================================================
// Kernel A: per-EDU attention pooling + projection.  grid=512, block=256.
// =========================================================================
__global__ __launch_bounds__(256) void edu_pool_kernel(
    const float* __restrict__ seq_all, const int* __restrict__ amask,
    const float* __restrict__ w1, const float* __restrict__ b1,
    const float* __restrict__ w2, const float* __restrict__ b2,
    const float* __restrict__ pw, const float* __restrict__ pb,
    float* __restrict__ enc_out)
{
  __shared__ alignas(32) __bf16 ldsA[2][8 * 2 * 512];   // 32 KB, frag-swizzled
  __shared__ alignas(32) __bf16 ldsB[2][7 * 2 * 512];   // 28 KB, frag-swizzled
  __shared__ float sc[128];
  __shared__ float wgt[128];
  __shared__ float pooled[HB];
  __shared__ float red[16];

  const int n    = blockIdx.x;
  const int t    = threadIdx.x;
  const int lane = t & 31;
  const int wv   = t >> 5;              // wave id == M-tile id (0..7)

  const float* seqn = seq_all + (size_t)n * SEQ * HB + HB;  // skip token 0

  if (t < 128) sc[t] = 0.f;

  v8f acc[7];
  #pragma unroll
  for (int i = 0; i < 7; ++i) acc[i] = {};

  // ---- GEMM: (128x768) @ (768x112) bf16/f32, double-buffered pipeline ----
  float4 ra[8], rb[7];
  load_regs(seqn, w1, 0, t, ra, rb);
  store_lds(ldsA[0], ldsB[0], t, ra, rb);
  __syncthreads();

  for (int kc = 0; kc < NKC; ++kc) {
    const int cur = kc & 1;
    if (kc + 1 < NKC)                       // global loads for next chunk
      load_regs(seqn, w1, (kc + 1) * KC, t, ra, rb);
    compute_chunk(ldsA[cur], ldsB[cur], lane, wv, acc);
    if (kc + 1 < NKC)                       // fill the other buffer
      store_lds(ldsA[cur ^ 1], ldsB[cur ^ 1], t, ra, rb);
    __syncthreads();
  }

  // ---- scores: sc[m] = sum_a tanh(a1pre + b1[a]) * w2[a] ----
  const int ncol = lane & 15;
  const int mofs = (lane < 16) ? 0 : 8;
  #pragma unroll
  for (int nt = 0; nt < 7; ++nt) {
    int nc = nt * 16 + ncol;
    bool nvalid = (nc < A_DIM);
    float w2v = nvalid ? w2[nc] : 0.f;
    float b1v = nvalid ? b1[nc] : 0.f;
    #pragma unroll
    for (int r = 0; r < 8; ++r) {
      int m = wv * 16 + mofs + r;
      if (nvalid && m < SROWS) {
        float a1 = tanhf(acc[nt][r] + b1v);
        atomicAdd(&sc[m], a1 * w2v);
      }
    }
  }
  __syncthreads();

  // ---- masked softmax over 127 tokens (mask multiplies scores, as ref) ----
  float b2v = b2[0];
  if (t < SROWS) wgt[t] = (sc[t] + b2v) * (float)amask[n * SEQ + 1 + t];
  __syncthreads();
  float mval = (t < SROWS) ? wgt[t] : -3.0e38f;
  #pragma unroll
  for (int off = 16; off > 0; off >>= 1) mval = fmaxf(mval, __shfl_down(mval, off));
  if (lane == 0) red[wv] = mval;
  __syncthreads();
  if (t == 0) {
    float m0 = red[0];
    for (int i = 1; i < 8; ++i) m0 = fmaxf(m0, red[i]);
    red[8] = m0;
  }
  __syncthreads();
  const float mx = red[8];
  float ev = (t < SROWS) ? __expf(wgt[t] - mx) : 0.f;
  float sval = ev;
  #pragma unroll
  for (int off = 16; off > 0; off >>= 1) sval += __shfl_down(sval, off);
  if (lane == 0) red[wv] = sval;
  __syncthreads();
  if (t == 0) {
    float s0 = 0.f;
    for (int i = 0; i < 8; ++i) s0 += red[i];
    red[9] = 1.f / s0;
  }
  __syncthreads();
  if (t < SROWS) wgt[t] = ev * red[9];
  __syncthreads();

  // ---- pooled[h] = sum_s seq[s][h] * wgt[s]  (3 h-values per thread) ----
  float p0 = 0.f, p1 = 0.f, p2 = 0.f;
  for (int s = 0; s < SROWS; ++s) {
    float w = wgt[s];
    const float* row = seqn + s * HB;
    p0 += row[t] * w;
    p1 += row[t + 256] * w;
    p2 += row[t + 512] * w;
  }
  pooled[t] = p0; pooled[t + 256] = p1; pooled[t + 512] = p2;
  __syncthreads();

  // ---- enc = pooled @ project_w + project_b  (one output channel/thread) ----
  float ev2 = pb[t];
  #pragma unroll 4
  for (int h = 0; h < HB; ++h) ev2 += pooled[h] * pw[h * H_DIM + t];
  enc_out[n * H_DIM + t] = ev2;
}

// =========================================================================
// Kernel B: sequential shift-reduce parser, 2N-1 steps.  grid=1, block=768.
// =========================================================================
__global__ __launch_bounds__(768) void parser_kernel(
    const float* __restrict__ enc, const float* __restrict__ missing,
    const float* __restrict__ aw, const float* __restrict__ ab,
    const float* __restrict__ tw, const float* __restrict__ tb,
    float* __restrict__ stack, float* __restrict__ out)
{
  __shared__ float feat[768];        // [s1 | s0 | b]
  __shared__ float g[640];
  __shared__ float merged[256];
  __shared__ float s2[2];
  __shared__ int sp_s, bi_s, shift_s;

  const int t = threadIdx.x;
  const int lane = t & 31;
  if (t == 0) { sp_s = 0; bi_s = 0; }
  __syncthreads();

  for (int step = 0; step < 2 * N_EDUS - 1; ++step) {
    const int sp = sp_s, bi = bi_s;
    if (t < 256) {
      feat[t] = (sp >= 2) ? stack[(sp - 2) * H_DIM + t] : missing[t];
    } else if (t < 512) {
      int j = t - 256;
      feat[t] = (sp >= 1) ? stack[(sp - 1) * H_DIM + j] : missing[j];
    } else {
      int j = t - 512;
      int bidx = (bi < N_EDUS) ? bi : (N_EDUS - 1);
      feat[t] = (bi < N_EDUS) ? enc[bidx * H_DIM + j] : missing[j];
    }
    if (t < 2) s2[t] = ab[t];
    __syncthreads();

    // scores = feat @ action_w + action_b  (wave reduce + LDS atomics)
    float f = feat[t];
    float q0 = f * aw[t * 2 + 0];
    float q1 = f * aw[t * 2 + 1];
    #pragma unroll
    for (int off = 16; off > 0; off >>= 1) {
      q0 += __shfl_down(q0, off);
      q1 += __shfl_down(q1, off);
    }
    if (lane == 0) { atomicAdd(&s2[0], q0); atomicAdd(&s2[1], q1); }

    // TreeLSTM gates: g = [h1;h2] @ tree_w + tree_b   (4 independent chains)
    if (t < 640) {
      float g0 = 0.f, g1 = 0.f, g2 = 0.f, g3 = 0.f;
      #pragma unroll 2
      for (int h = 0; h < HALF; h += 4) {                 // h1 = feat[0:128]
        g0 += feat[h + 0] * tw[(h + 0) * 640 + t];
        g1 += feat[h + 1] * tw[(h + 1) * 640 + t];
        g2 += feat[h + 2] * tw[(h + 2) * 640 + t];
        g3 += feat[h + 3] * tw[(h + 3) * 640 + t];
      }
      #pragma unroll 2
      for (int h = 0; h < HALF; h += 4) {                 // h2 = feat[256:384]
        g0 += feat[256 + h + 0] * tw[(HALF + h + 0) * 640 + t];
        g1 += feat[256 + h + 1] * tw[(HALF + h + 1) * 640 + t];
        g2 += feat[256 + h + 2] * tw[(HALF + h + 2) * 640 + t];
        g3 += feat[256 + h + 3] * tw[(HALF + h + 3) * 640 + t];
      }
      g[t] = tb[t] + (g0 + g1) + (g2 + g3);
    }
    __syncthreads();

    if (t == 0) {
      float sc0 = (bi < N_EDUS) ? s2[0] : -1.0e9f;
      float sc1 = (sp >= 2) ? s2[1] : -1.0e9f;
      shift_s = (sc0 >= sc1) ? 1 : 0;   // argmax ties -> index 0 (shift)
    }
    if (t < HALF) {
      float gi = g[t], f1 = g[HALF + t], f2 = g[2 * HALF + t];
      float go = g[3 * HALF + t], gu = g[4 * HALF + t];
      float c1 = feat[HALF + t];         // s1 cell
      float c2 = feat[256 + HALF + t];   // s0 cell
      float si  = 1.f / (1.f + __expf(-gi));
      float sf1 = 1.f / (1.f + __expf(-f1));
      float sf2 = 1.f / (1.f + __expf(-f2));
      float so  = 1.f / (1.f + __expf(-go));
      float c = si * tanhf(gu) + sf1 * c1 + sf2 * c2;
      float h = so * tanhf(c);
      merged[t] = h;
      merged[HALF + t] = c;
    }
    __syncthreads();

    const int shift = shift_s;
    const int nidx = shift ? sp : ((sp >= 2) ? sp - 2 : 0);
    if (t < 256) {
      float nv = shift ? feat[512 + t] : merged[t];
      stack[nidx * H_DIM + t] = nv;
    }
    if (t == 0) {
      sp_s = shift ? sp + 1 : sp - 1;
      bi_s = shift ? bi + 1 : bi;
    }
    __syncthreads();
  }
  if (t < 256) out[t] = stack[t];
}

// =========================================================================
extern "C" void kernel_launch(void* const* d_in, const int* in_sizes, int n_in,
                              void* d_out, int out_size, void* d_ws, size_t ws_size,
                              hipStream_t stream) {
  const float* seq  = (const float*)d_in[0];
  const int*   am   = (const int*)  d_in[1];
  const float* w1   = (const float*)d_in[2];
  const float* b1   = (const float*)d_in[3];
  const float* w2   = (const float*)d_in[4];
  const float* b2   = (const float*)d_in[5];
  const float* pw   = (const float*)d_in[6];
  const float* pb   = (const float*)d_in[7];
  const float* miss = (const float*)d_in[8];
  const float* aw   = (const float*)d_in[9];
  const float* ab   = (const float*)d_in[10];
  const float* tw   = (const float*)d_in[11];
  const float* tb   = (const float*)d_in[12];

  float* enc   = (float*)d_ws;                 // 512*256 floats
  float* stack = enc + N_EDUS * H_DIM;         // 512*256 floats

  edu_pool_kernel<<<N_EDUS, 256, 0, stream>>>(seq, am, w1, b1, w2, b2, pw, pb, enc);
  parser_kernel<<<1, 768, 0, stream>>>(enc, miss, aw, ab, tw, tb, stack, (float*)d_out);
}
